// Net_41266045780423
// MI455X (gfx1250) — matrix operand order repaired and checked by reading, not compiled, not run on hardware
//
#include <hip/hip_runtime.h>

typedef __attribute__((ext_vector_type(16))) _Float16 v16h;
typedef __attribute__((ext_vector_type(8)))  float    v8f;

#define DEVFN static __device__ __forceinline__

DEVFN v8f wmma_f16(v16h a, v16h b, v8f c) {
  // D = A(16x32 f16) * B(32x16 f16) + C(16x16 f32)
  return __builtin_amdgcn_wmma_f32_16x16x32_f16(false, a, false, b, (short)0, c,
                                                false, false);
}

// A fragment from row-major [m][k] f16, stride in elements (mult of 8, 16B-aligned).
// ISA layout (16-bit A 16x32): lane L holds row M=L&15; low lanes K {0..7,16..23},
// high lanes +8. => two contiguous b128 loads per lane.
DEVFN v16h load_a(const _Float16* base, int stride, int m0, int k0) {
  int lane = threadIdx.x & 31;
  const _Float16* p = base + (size_t)(m0 + (lane & 15)) * stride + k0 + ((lane >> 4) << 3);
  v16h r;
  *(float4*)&r       = *(const float4*)p;        // K = kb .. kb+7
  *((float4*)&r + 1) = *(const float4*)(p + 16); // K = kb+16 .. kb+23
  return r;
}

// B fragment from weights stored [n][k] row-major f16 (x @ W^T convention).
// ISA layout (16-bit B 32x16): lane L holds col N=L&15; low lanes K {0..7,16..23},
// high lanes +8. => two contiguous b128 loads.
DEVFN v16h load_b(const _Float16* base, int stride, int n0, int k0) {
  int lane = threadIdx.x & 31;
  const _Float16* p = base + (size_t)(n0 + (lane & 15)) * stride + k0 + ((lane >> 4) << 3);
  v16h r;
  *(float4*)&r       = *(const float4*)p;        // K = kb .. kb+7
  *((float4*)&r + 1) = *(const float4*)(p + 16); // K = kb+16 .. kb+23
  return r;
}

// ---------------- weight repack: fp32 [N][K] -> f16 [Np][Kp] zero-padded -------------
__global__ void repack_nk(const float* __restrict__ src, _Float16* __restrict__ dst,
                          int N, int K, int Np, int Kp) {
  int i = blockIdx.x * blockDim.x + threadIdx.x;
  if (i >= Np * Kp) return;
  int n = i / Kp, k = i % Kp;
  float v = (n < N && k < K) ? src[n * K + k] : 0.f;
  dst[i] = (_Float16)v;
}

// fp32 [D][H] -> f16 [Hp][Dp] transposed+padded (for einsum 'bd,edh' experts)
__global__ void repack_tr(const float* __restrict__ src, _Float16* __restrict__ dst,
                          int D, int H, int Dp, int Hp) {
  int i = blockIdx.x * blockDim.x + threadIdx.x;
  if (i >= Hp * Dp) return;
  int h = i / Dp, d = i % Dp;
  float v = (h < H && d < D) ? src[d * H + h] : 0.f;
  dst[i] = (_Float16)v;
}

__global__ void pad_bias(const float* __restrict__ src, float* __restrict__ dst,
                         int N, int Np) {
  int i = blockIdx.x * blockDim.x + threadIdx.x;
  if (i < Np) dst[i] = (i < N) ? src[i] : 0.f;
}

// ---------------- conv1 (implicit GEMM, K=75->96) + ReLU + 2x2 maxpool ---------------
// All K indices are compile-time per lane-half => every gather is a single
// global_load_b32 with an immediate offset from one per-lane base pointer.
template <int LH>
DEVFN void conv1_gather(const float* __restrict__ p0, _Float16 av[3][16]) {
#pragma unroll
  for (int kc = 0; kc < 3; ++kc) {
#pragma unroll
    for (int i = 0; i < 16; ++i) {
      const int K = kc * 32 + i + ((i >= 8) ? 8 : 0) + LH * 8;
      float v = 0.f;
      if (K < 75) {  // folds at compile time
        const int off = (K / 25) * 1024 + ((K % 25) / 5) * 32 + (K % 25 % 5);
        v = p0[off];
      }
      av[kc][i] = (_Float16)v;
    }
  }
}

__global__ __launch_bounds__(256) void conv1_kernel(
    const float* __restrict__ x, const _Float16* __restrict__ W1,
    const float* __restrict__ b1, _Float16* __restrict__ h1) {
  int wid  = (blockIdx.x * blockDim.x + threadIdx.x) >> 5;
  int lane = threadIdx.x & 31;
  int b = wid / 49, t = wid % 49;
  int lh = lane >> 4;
  int M = lane & 15;
  int pp = t * 4 + (M >> 2);              // pooled flat position (0..195)
  int q  = M & 3;
  int iy0 = (pp / 14) * 2 + (q >> 1);     // pre-pool conv output row (0..27)
  int ix0 = (pp % 14) * 2 + (q & 1);
  const float* p0 = x + (size_t)b * 3 * 32 * 32 + iy0 * 32 + ix0;

  _Float16 av[3][16];
  if (lh == 0) conv1_gather<0>(p0, av);   // divergent loads only; WMMA outside
  else         conv1_gather<1>(p0, av);

  v8f acc = {};
#pragma unroll
  for (int kc = 0; kc < 3; ++kc) {
    v16h a;
#pragma unroll
    for (int i = 0; i < 16; ++i) a[i] = av[kc][i];
    acc = wmma_f16(a, load_b(W1, 96, 0, kc * 32), acc);
  }

  int n = lane & 15;
  float bias = b1[n];
#pragma unroll
  for (int p_ = 0; p_ < 2; ++p_) {
    float m4 = fmaxf(fmaxf(acc[4 * p_ + 0], acc[4 * p_ + 1]),
                     fmaxf(acc[4 * p_ + 2], acc[4 * p_ + 3]));
    float v = fmaxf(m4 + bias, 0.f);
    int ppo = t * 4 + 2 * lh + p_;
    if (n < 6) h1[(size_t)(b * 6 + n) * 196 + ppo] = (_Float16)v;
  }
}

// ---------------- conv2 (K=150->160) + ReLU + pool, writes h2 [B][416] f16 -----------
template <int LH>
DEVFN void conv2_gather(const _Float16* __restrict__ p0, bool valid, _Float16 av[5][16]) {
#pragma unroll
  for (int kc = 0; kc < 5; ++kc) {
#pragma unroll
    for (int i = 0; i < 16; ++i) {
      const int K = kc * 32 + i + ((i >= 8) ? 8 : 0) + LH * 8;
      _Float16 v = (_Float16)0.f;
      if (K < 150) {  // folds at compile time
        const int off = (K / 25) * 196 + ((K % 25) / 5) * 14 + (K % 25 % 5);
        if (valid) v = p0[off];
      }
      av[kc][i] = v;
    }
  }
}

__global__ __launch_bounds__(256) void conv2_kernel(
    const _Float16* __restrict__ h1, const _Float16* __restrict__ W2,
    const float* __restrict__ b2, _Float16* __restrict__ h2) {
  int wid  = (blockIdx.x * blockDim.x + threadIdx.x) >> 5;
  int lane = threadIdx.x & 31;
  int b = wid / 7, t = wid % 7;
  int lh = lane >> 4;
  int M = lane & 15;
  int pp = t * 4 + (M >> 2);              // pooled flat (valid < 25)
  int q  = M & 3;
  int iy0 = (pp / 5) * 2 + (q >> 1);      // pre-pool conv output row (0..9)
  int ix0 = (pp % 5) * 2 + (q & 1);
  const _Float16* p0 = h1 + (size_t)b * 6 * 196 + iy0 * 14 + ix0;
  bool valid = pp < 25;

  _Float16 av[5][16];
  if (lh == 0) conv2_gather<0>(p0, valid, av);
  else         conv2_gather<1>(p0, valid, av);

  v8f acc = {};
#pragma unroll
  for (int kc = 0; kc < 5; ++kc) {
    v16h a;
#pragma unroll
    for (int i = 0; i < 16; ++i) a[i] = av[kc][i];
    acc = wmma_f16(a, load_b(W2, 160, 0, kc * 32), acc);
  }

  int n = lane & 15;
  float bias = b2[n];
#pragma unroll
  for (int p_ = 0; p_ < 2; ++p_) {
    float m4 = fmaxf(fmaxf(acc[4 * p_ + 0], acc[4 * p_ + 1]),
                     fmaxf(acc[4 * p_ + 2], acc[4 * p_ + 3]));
    float v = fmaxf(m4 + bias, 0.f);
    int ppo = t * 4 + 2 * lh + p_;
    if (ppo < 25) h2[(size_t)b * 416 + n * 25 + ppo] = (_Float16)v;  // [16,5,5] flatten
  }
  if (t == 0 && lane < 16) h2[(size_t)b * 416 + 400 + lane] = (_Float16)0.f;  // K pad
}

// ---------------- fc1: [B,416]x[128,416]^T, 13 K-chunks, ReLU -> h3 [B][128] ---------
__global__ __launch_bounds__(256) void fc1_kernel(
    const _Float16* __restrict__ h2, const _Float16* __restrict__ W3,
    const float* __restrict__ b3, _Float16* __restrict__ h3) {
  int wid  = (blockIdx.x * blockDim.x + threadIdx.x) >> 5;
  int lane = threadIdx.x & 31;
  int nt = wid & 7, m0 = (wid >> 3) * 16;
  v8f acc = {};
  for (int kc = 0; kc < 13; ++kc)
    acc = wmma_f16(load_a(h2, 416, m0, kc * 32), load_b(W3, 416, nt * 16, kc * 32), acc);
  int lh = lane >> 4;
  int n = nt * 16 + (lane & 15);
  float bias = b3[n];
#pragma unroll
  for (int r = 0; r < 8; ++r) {
    float v = fmaxf(acc[r] + bias, 0.f);
    h3[(size_t)(m0 + r + 8 * lh) * 128 + n] = (_Float16)v;
  }
}

// ------- fused: fc2+ReLU -> gate softmax/argmax -> both experts -> select -> fc4 -----
__global__ __launch_bounds__(128) void moe_tail_kernel(
    const _Float16* __restrict__ h3, const _Float16* __restrict__ W4,
    const float* __restrict__ b4, const float* __restrict__ gate_w,
    const _Float16* __restrict__ WE, const float* __restrict__ bE,
    const _Float16* __restrict__ W5, const float* __restrict__ fc4_b,
    float* __restrict__ out) {
  __shared__ __align__(16) _Float16 s_h[4][16][96];
  __shared__ __align__(16) _Float16 s_sel[4][16][96];
  __shared__ float s_g[4][16];
  __shared__ int   s_idx[4][16];

  int w    = threadIdx.x >> 5;
  int lane = threadIdx.x & 31;
  int lh   = lane >> 4;
  int m0   = (blockIdx.x * 4 + w) * 16;

  // fc2: [16,128] x [96,128]^T, ReLU, stage rows into LDS
  for (int nt = 0; nt < 6; ++nt) {
    v8f acc = {};
#pragma unroll
    for (int kc = 0; kc < 4; ++kc)
      acc = wmma_f16(load_a(h3, 128, m0, kc * 32), load_b(W4, 128, nt * 16, kc * 32), acc);
    int n = nt * 16 + (lane & 15);
    float bias = b4[n];
#pragma unroll
    for (int r = 0; r < 8; ++r)
      s_h[w][r + 8 * lh][n] = (_Float16)fmaxf(acc[r] + bias, 0.f);
  }
  __syncthreads();

  // gating (N=2: VALU). gate = softmax max prob = 1/(1+exp(-|l0-l1|))
  if (lane < 16) {
    float l0 = 0.f, l1 = 0.f;
    for (int d = 0; d < 84; ++d) {
      float hv = (float)s_h[w][lane][d];
      l0 += hv * gate_w[d];
      l1 += hv * gate_w[84 + d];
    }
    int   e  = (l1 > l0) ? 1 : 0;                 // argmax, first-on-tie
    float ad = fabsf(l0 - l1);
    s_g[w][lane]   = 1.f / (1.f + __expf(-ad));
    s_idx[w][lane] = e;
  }
  __syncthreads();

  // both experts densely via WMMA; per-token select + gate scale into s_sel
  const _Float16* s_hp = &s_h[w][0][0];
  for (int nt = 0; nt < 6; ++nt) {
    v8f a0 = {}, a1 = {};
#pragma unroll
    for (int kc = 0; kc < 3; ++kc) {
      v16h av = load_a(s_hp, 96, 0, kc * 32);
      a0 = wmma_f16(av, load_b(WE,           96, nt * 16, kc * 32), a0);
      a1 = wmma_f16(av, load_b(WE + 96 * 96, 96, nt * 16, kc * 32), a1);
    }
    int n = nt * 16 + (lane & 15);
#pragma unroll
    for (int r = 0; r < 8; ++r) {
      int tl = r + 8 * lh;
      int e  = s_idx[w][tl];
      float v = (e ? a1[r] : a0[r]) + bE[e * 96 + n];
      s_sel[w][tl][n] = (_Float16)(v * s_g[w][tl]);
    }
  }
  __syncthreads();

  // fc4: [16,96] x [16,96]^T (10 valid rows), K=3 chunks
  const _Float16* s_sp = &s_sel[w][0][0];
  v8f acc = {};
#pragma unroll
  for (int kc = 0; kc < 3; ++kc)
    acc = wmma_f16(load_a(s_sp, 96, 0, kc * 32), load_b(W5, 96, 0, kc * 32), acc);
  int n = lane & 15;
  if (n < 10) {
    float bias = fc4_b[n];
#pragma unroll
    for (int r = 0; r < 8; ++r)
      out[(size_t)(m0 + r + 8 * lh) * 10 + n] = acc[r] + bias;
  }
}

// ------------------------------------ launch -----------------------------------------
extern "C" void kernel_launch(void* const* d_in, const int* in_sizes, int n_in,
                              void* d_out, int out_size, void* d_ws, size_t ws_size,
                              hipStream_t stream) {
  const float* x        = (const float*)d_in[0];
  const float* conv1_w  = (const float*)d_in[1];
  const float* conv1_b  = (const float*)d_in[2];
  const float* conv2_w  = (const float*)d_in[3];
  const float* conv2_b  = (const float*)d_in[4];
  const float* fc1_w    = (const float*)d_in[5];
  const float* fc1_b    = (const float*)d_in[6];
  const float* fc2_w    = (const float*)d_in[7];
  const float* fc2_b    = (const float*)d_in[8];
  const float* gate_w   = (const float*)d_in[9];
  const float* expert_w = (const float*)d_in[10];
  const float* expert_b = (const float*)d_in[11];
  const float* fc4_w    = (const float*)d_in[12];
  const float* fc4_b    = (const float*)d_in[13];
  float* out = (float*)d_out;

  char* ws = (char*)d_ws;
  // packed f16 weights ([Npad][Kpad])
  _Float16* W1 = (_Float16*)(ws + 0);         // [16][96]
  _Float16* W2 = (_Float16*)(ws + 4096);      // [16][160]
  _Float16* W3 = (_Float16*)(ws + 16384);     // [128][416]
  _Float16* W4 = (_Float16*)(ws + 131072);    // [96][128]
  _Float16* WE = (_Float16*)(ws + 163840);    // [2][96][96]
  _Float16* W5 = (_Float16*)(ws + 204800);    // [16][96]
  float* b1 = (float*)(ws + 212992);          // [16]
  float* b2 = (float*)(ws + 213504);          // [16]
  float* b3 = (float*)(ws + 214016);          // [128]
  float* b4 = (float*)(ws + 214528);          // [96]
  float* bE = (float*)(ws + 215040);          // [2][96]
  // f16 activations
  _Float16* h1 = (_Float16*)(ws + 262144);    // [B][6][14][14]  38.5 MB
  _Float16* h2 = (_Float16*)(ws + 38797312);  // [B][416]        13.6 MB
  _Float16* h3 = (_Float16*)(ws + 52428800);  // [B][128]         4.2 MB

  const int B = 16384;

  // weight / bias repack prologue (tiny)
  repack_nk<<<(16 * 96 + 255) / 256, 256, 0, stream>>>(conv1_w, W1, 6, 75, 16, 96);
  repack_nk<<<(16 * 160 + 255) / 256, 256, 0, stream>>>(conv2_w, W2, 16, 150, 16, 160);
  repack_nk<<<(128 * 416 + 255) / 256, 256, 0, stream>>>(fc1_w, W3, 120, 400, 128, 416);
  repack_nk<<<(96 * 128 + 255) / 256, 256, 0, stream>>>(fc2_w, W4, 84, 120, 96, 128);
  repack_tr<<<(96 * 96 + 255) / 256, 256, 0, stream>>>(expert_w,           WE,           84, 84, 96, 96);
  repack_tr<<<(96 * 96 + 255) / 256, 256, 0, stream>>>(expert_w + 84 * 84, WE + 96 * 96, 84, 84, 96, 96);
  repack_nk<<<(16 * 96 + 255) / 256, 256, 0, stream>>>(fc4_w, W5, 10, 84, 16, 96);
  pad_bias<<<1, 64, 0, stream>>>(conv1_b, b1, 6, 16);
  pad_bias<<<1, 64, 0, stream>>>(conv2_b, b2, 16, 16);
  pad_bias<<<1, 128, 0, stream>>>(fc1_b, b3, 120, 128);
  pad_bias<<<1, 96, 0, stream>>>(fc2_b, b4, 84, 96);
  pad_bias<<<1, 96, 0, stream>>>(expert_b, bE, 84, 96);
  pad_bias<<<1, 96, 0, stream>>>(expert_b + 84, bE + 96, 84, 96);

  // pipeline (intermediates live in L2)
  conv1_kernel<<<B * 49 / 8, 256, 0, stream>>>(x, W1, b1, h1);   // 100352 blocks
  conv2_kernel<<<B * 7 / 8, 256, 0, stream>>>(h1, W2, b2, h2);   // 14336 blocks
  fc1_kernel<<<(B / 16) * 8 / 8, 256, 0, stream>>>(h2, W3, b3, h3); // 1024 blocks
  moe_tail_kernel<<<B / 16 / 4, 128, 0, stream>>>(h3, W4, b4, gate_w, WE, bE, W5,
                                                  fc4_b, out);   // 256 blocks
}
